// GNN1_36773509988938
// MI455X (gfx1250) — compile-verified, matrix operand order checked by reading, not run on hardware
//
#include <hip/hip_runtime.h>
#include <cstdint>
#include <cstddef>

// ---------------------------------------------------------------------------
// GNN1 forward for MI455X (gfx1250): WMMA GEMMs + atomic scatter softmax.
// ---------------------------------------------------------------------------

namespace {

constexpr int   NN     = 50000;
constexpr int   EE     = 800000;
constexpr int   IN_DIM = 100;
constexpr int   DD     = 128;
constexpr int   DT     = 32;
constexpr int   NOUTR  = 4278;
constexpr float BN_EPS = 1e-5f;

typedef __attribute__((ext_vector_type(16))) _Float16 v16h;
typedef __attribute__((ext_vector_type(8)))  float    v8f;

// ---------------- small utility kernels ----------------

__global__ void k_fill_u32(unsigned* __restrict__ p, unsigned v, size_t n) {
    size_t i = (size_t)blockIdx.x * blockDim.x + threadIdx.x;
    if (i < n) p[i] = v;
}

// h0[n][0:100]=x, [100:132]=node_emb[node_type], [132:160]=0
__global__ void k_build_h0(const float* __restrict__ x, const int* __restrict__ ntype,
                           const float* __restrict__ nemb, float* __restrict__ h0) {
    size_t gid = (size_t)blockIdx.x * blockDim.x + threadIdx.x;
    if (gid >= (size_t)NN * 160) return;
    int n = (int)(gid / 160), c = (int)(gid % 160);
    float v;
    if (c < IN_DIM)            v = x[(size_t)n * IN_DIM + c];
    else if (c < IN_DIM + DT)  v = nemb[ntype[n] * DT + (c - IN_DIM)];
    else                       v = 0.f;
    h0[gid] = v;
}

// scatter-mean of edge embeddings onto dst nodes (self-loop attr)
__global__ void k_loop_scatter(const int* __restrict__ dst, const int* __restrict__ etype,
                               const float* __restrict__ eemb,
                               float* __restrict__ lat, float* __restrict__ cnt) {
    size_t gid = (size_t)blockIdx.x * blockDim.x + threadIdx.x;
    if (gid >= (size_t)EE * DT) return;
    int e = (int)(gid >> 5), c = (int)(gid & 31);
    int d = dst[e], t = etype[e];
    atomicAdd(&lat[(size_t)d * DT + c], eemb[t * DT + c]);
    if (c == 0) atomicAdd(&cnt[d], 1.0f);
}

__global__ void k_loop_fin(float* __restrict__ lat, const float* __restrict__ cnt) {
    size_t gid = (size_t)blockIdx.x * blockDim.x + threadIdx.x;
    if (gid >= (size_t)NN * DT) return;
    int n = (int)(gid >> 5);
    lat[gid] /= fmaxf(cnt[n], 1.0f);
}

__global__ void k_init_bias(float* __restrict__ out, const float* __restrict__ bias) {
    size_t gid = (size_t)blockIdx.x * blockDim.x + threadIdx.x;
    if (gid >= (size_t)NN * DD) return;
    out[gid] = bias[gid & (DD - 1)];
}

// per-column sum & sumsq (for BatchNorm folding)
__global__ void k_bn_stats(const float* __restrict__ X, int M, int C, int ld,
                           float* __restrict__ sums, int rowsPerBlock) {
    int c = threadIdx.x;
    if (c >= C) return;
    int r0 = blockIdx.x * rowsPerBlock;
    int r1 = r0 + rowsPerBlock; if (r1 > M) r1 = M;
    float s = 0.f, q = 0.f;
    for (int r = r0; r < r1; ++r) {
        float v = X[(size_t)r * ld + c];
        s += v; q += v * v;
    }
    atomicAdd(&sums[c], s);
    atomicAdd(&sums[C + c], q);
}

// Fold BN into following Linear: Wf[i][j] = W[i][j]*g[j]/sigma[j] (zero-pad to Kpad)
//                                bf[i]    = blin[i] + sum_j W[i][j]*(beta[j]-mu[j]*g[j]/sigma[j])
__global__ void k_bn_fold(const float* __restrict__ W, const float* __restrict__ g,
                          const float* __restrict__ beta, const float* __restrict__ blin,
                          const float* __restrict__ sums, float invM,
                          float* __restrict__ Wf, float* __restrict__ bf,
                          int Kin, int Kpad) {
    int i = blockIdx.x;                     // output row (0..Nout-1)
    __shared__ float red[256];
    float part = 0.f;
    for (int j = threadIdx.x; j < Kpad; j += blockDim.x) {
        float wij = 0.f, sc = 0.f, cc = 0.f;
        if (j < Kin) {
            wij = W[(size_t)i * Kin + j];
            float mean = sums[j] * invM;
            float var  = sums[Kin + j] * invM - mean * mean;
            float istd = rsqrtf(var + BN_EPS);
            sc = g[j] * istd;
            cc = beta[j] - mean * sc;
        }
        Wf[(size_t)i * Kpad + j] = wij * sc;
        part += wij * cc;
    }
    red[threadIdx.x] = part;
    __syncthreads();
    for (int s = 128; s > 0; s >>= 1) {
        if ((int)threadIdx.x < s) red[threadIdx.x] += red[threadIdx.x + s];
        __syncthreads();
    }
    if (threadIdx.x == 0) bf[i] = blin[i] + red[0];
}

// ---------------- WMMA GEMM: Y = X @ W^T (+bias) (+lrelu) (+=) ----------------
// One wave computes a 16 x (16*CT) output slab: A loaded once per K-step,
// CT back-to-back v_wmma_f32_16x16x32_f16 issues. Row/col indices are clamped
// instead of predicated: out-of-range A rows / B cols only feed D elements
// that are never stored, so the K-loop is fully branch-free (EXEC stays all-1s).

__device__ __forceinline__ void load8h(const float* __restrict__ p, v16h& v, int base) {
    const float4* q = (const float4*)p;
    float4 a = q[0], b = q[1];
    v[base + 0] = (_Float16)a.x; v[base + 1] = (_Float16)a.y;
    v[base + 2] = (_Float16)a.z; v[base + 3] = (_Float16)a.w;
    v[base + 4] = (_Float16)b.x; v[base + 5] = (_Float16)b.y;
    v[base + 6] = (_Float16)b.z; v[base + 7] = (_Float16)b.w;
}

template<int K, int CT>
__global__ void k_gemm_wmma(const float* __restrict__ X, int ldx,
                            const float* __restrict__ W, int ldw, int Nw,
                            const float* __restrict__ bias,
                            float* __restrict__ Y, int ldy,
                            int M, int Nout, int nGroups,
                            float slope, int doLrelu, int accMode) {
    const int wave = threadIdx.x >> 5;
    const int lane = threadIdx.x & 31;
    const int nRowTiles = (M + 15) >> 4;
    const int tile = blockIdx.x * (blockDim.x >> 5) + wave;
    if (tile >= nRowTiles * nGroups) return;    // uniform per wave: EXEC stays all-ones
    const int tm  = tile / nGroups;
    const int tn0 = (tile % nGroups) * CT;
    const int l   = lane & 15;
    const int hi  = lane >> 4;

    int arow = tm * 16 + l; if (arow >= M) arow = M - 1;          // clamp, no branch
    const float* Ap = X + (size_t)arow * ldx + hi * 8;
    const float* Bp[CT];
    #pragma unroll
    for (int ct = 0; ct < CT; ++ct) {
        int brow = (tn0 + ct) * 16 + l; if (brow >= Nw) brow = Nw - 1;
        Bp[ct] = W + (size_t)brow * ldw + hi * 8;
    }

    v8f zero = {};
    v8f acc[CT];
    #pragma unroll
    for (int ct = 0; ct < CT; ++ct) acc[ct] = zero;

    #pragma unroll
    for (int kb = 0; kb < K; kb += 32) {
        v16h a;
        // 16-bit A layout: lanes 0-15 rows, K {kb..kb+7, kb+16..kb+23};
        //                  lanes 16-31 rows, K {kb+8..kb+15, kb+24..kb+31}
        load8h(Ap + kb,      a, 0);
        load8h(Ap + kb + 16, a, 8);
        #pragma unroll
        for (int ct = 0; ct < CT; ++ct) {
            v16h b;
            load8h(Bp[ct] + kb,      b, 0);   // B mirrored by column: B[k][n] = W[n][k]
            load8h(Bp[ct] + kb + 16, b, 8);
            acc[ct] = __builtin_amdgcn_wmma_f32_16x16x32_f16(
                false, a, false, b, (short)0, acc[ct], false, false);
        }
    }

    #pragma unroll
    for (int ct = 0; ct < CT; ++ct) {
        const int   n  = (tn0 + ct) * 16 + l;
        const float bb = (bias != nullptr && n < Nw) ? bias[n] : 0.f;
        #pragma unroll
        for (int v = 0; v < 8; ++v) {
            const int r = tm * 16 + hi * 8 + v;   // C/D layout: VGPR v -> M=v (+8 for hi lanes)
            if (r >= M) continue;
            float val = acc[ct][v] + bb;
            if (doLrelu) val = (val >= 0.f) ? val : slope * val;
            if (n < Nout) {
                if (accMode) Y[(size_t)r * ldy + n] += val;
                else         Y[(size_t)r * ldy + n]  = val;
            } else if (!accMode && n < ldy) {
                Y[(size_t)r * ldy + n] = 0.f;     // keep K-padding zeroed for next GEMM
            }
        }
    }
}

// ---------------- GATv2 edge kernels (scatter softmax) ----------------

__device__ __forceinline__ unsigned encf(float f) {
    unsigned u = __float_as_uint(f);
    return (u & 0x80000000u) ? ~u : (u | 0x80000000u);   // monotone order-preserving
}
__device__ __forceinline__ float decf(unsigned u) {
    unsigned v = (u & 0x80000000u) ? (u & 0x7FFFFFFFu) : ~u;
    return __uint_as_float(v);
}

// pass 1: per (edge, head) score + atomicMax into per-(dst,head) max
__global__ void k_gat_scores(const float* __restrict__ xl, const float* __restrict__ xr,
                             const float* __restrict__ esT, const float* __restrict__ lee,
                             const int* __restrict__ src, const int* __restrict__ dst,
                             const int* __restrict__ etype, const float* __restrict__ att,
                             float* __restrict__ sbuf, unsigned* __restrict__ smax) {
    size_t gid = (size_t)blockIdx.x * blockDim.x + threadIdx.x;
    size_t tot = (size_t)(EE + NN) * 8;
    if (gid >= tot) return;
    int h = (int)(gid & 7);
    long e = (long)(gid >> 3);
    int s, d; const float* ee;
    if (e < EE) { s = src[e]; d = dst[e]; ee = esT + etype[e] * DD; }
    else        { s = d = (int)(e - EE); ee = lee + (size_t)s * DD; }
    const float4* pl = (const float4*)(xl + (size_t)s * DD + h * 16);
    const float4* pr = (const float4*)(xr + (size_t)d * DD + h * 16);
    const float4* pe = (const float4*)(ee + h * 16);
    const float4* pa = (const float4*)(att + h * 16);
    float sc = 0.f;
    #pragma unroll
    for (int q = 0; q < 4; ++q) {
        float4 a4 = pl[q], r4 = pr[q], e4 = pe[q], t4 = pa[q];
        float m;
        m = a4.x + r4.x + e4.x; m = (m >= 0.f) ? m : 0.2f * m; sc += m * t4.x;
        m = a4.y + r4.y + e4.y; m = (m >= 0.f) ? m : 0.2f * m; sc += m * t4.y;
        m = a4.z + r4.z + e4.z; m = (m >= 0.f) ? m : 0.2f * m; sc += m * t4.z;
        m = a4.w + r4.w + e4.w; m = (m >= 0.f) ? m : 0.2f * m; sc += m * t4.w;
    }
    sbuf[gid] = sc;
    atomicMax(&smax[(size_t)d * 8 + h], encf(sc));
}

// pass 2: exp(s - max) and denominator
__global__ void k_gat_exp(const int* __restrict__ dst, float* __restrict__ sbuf,
                          const unsigned* __restrict__ smax, float* __restrict__ den) {
    size_t gid = (size_t)blockIdx.x * blockDim.x + threadIdx.x;
    size_t tot = (size_t)(EE + NN) * 8;
    if (gid >= tot) return;
    int h = (int)(gid & 7);
    long e = (long)(gid >> 3);
    int d = (e < EE) ? dst[e] : (int)(e - EE);
    float ex = __expf(sbuf[gid] - decf(smax[(size_t)d * 8 + h]));
    sbuf[gid] = ex;
    atomicAdd(&den[(size_t)d * 8 + h], ex);
}

// pass 3: out[dst] += alpha * xl[src]
__global__ void k_gat_agg(const float* __restrict__ xl, const int* __restrict__ src,
                          const int* __restrict__ dst, const float* __restrict__ sbuf,
                          const float* __restrict__ den, float* __restrict__ out) {
    size_t gid = (size_t)blockIdx.x * blockDim.x + threadIdx.x;
    size_t tot = (size_t)(EE + NN) * 8;
    if (gid >= tot) return;
    int h = (int)(gid & 7);
    long e = (long)(gid >> 3);
    int s, d;
    if (e < EE) { s = src[e]; d = dst[e]; }
    else        { s = d = (int)(e - EE); }
    float alpha = sbuf[gid] / den[(size_t)d * 8 + h];
    const float4* pl = (const float4*)(xl + (size_t)s * DD + h * 16);
    float*        po = out + (size_t)d * DD + h * 16;
    #pragma unroll
    for (int q = 0; q < 4; ++q) {
        float4 a4 = pl[q];
        atomicAdd(&po[q * 4 + 0], alpha * a4.x);
        atomicAdd(&po[q * 4 + 1], alpha * a4.y);
        atomicAdd(&po[q * 4 + 2], alpha * a4.z);
        atomicAdd(&po[q * 4 + 3], alpha * a4.w);
    }
}

// ---------------- host helpers ----------------

inline void zero_u32(hipStream_t st, void* p, size_t n32) {
    k_fill_u32<<<(unsigned)((n32 + 255) / 256), 256, 0, st>>>((unsigned*)p, 0u, n32);
}

inline void gemm(hipStream_t st, int K, const float* X, int ldx,
                 const float* W, int ldw, int Nw, const float* bias,
                 float* Y, int ldy, int M, int Nout, int nct,
                 float slope, int doL, int acc) {
    const int CT = (nct % 4 == 0) ? 4 : 2;
    const int nGroups = nct / CT;
    int tiles  = ((M + 15) / 16) * nGroups;
    int blocks = (tiles + 7) / 8;
    dim3 g(blocks), b(256);
#define GNN1_DISP(KK)                                                                         \
    if (CT == 4) k_gemm_wmma<KK, 4><<<g, b, 0, st>>>(X, ldx, W, ldw, Nw, bias, Y, ldy, M,     \
                                                     Nout, nGroups, slope, doL, acc);         \
    else         k_gemm_wmma<KK, 2><<<g, b, 0, st>>>(X, ldx, W, ldw, Nw, bias, Y, ldy, M,     \
                                                     Nout, nGroups, slope, doL, acc);
    switch (K) {
        case 32:  { GNN1_DISP(32)  } break;
        case 128: { GNN1_DISP(128) } break;
        case 160: { GNN1_DISP(160) } break;
        case 256: { GNN1_DISP(256) } break;
        default: break;
    }
#undef GNN1_DISP
}

struct MlpIdx { int g1, b1, W1, bb1, g2, b2, W2, bb2; };
struct BlkIdx { int Wl, bl, Wr, br, We, att, bias; MlpIdx mlp; };
struct IdxMap {
    int x, ei, nt, et, nemb, eemb;
    MlpIdx inp, out;
    BlkIdx blk[3];
};

} // namespace

extern "C" void kernel_launch(void* const* d_in, const int* in_sizes, int n_in,
                              void* d_out, int out_size, void* d_ws, size_t ws_size,
                              hipStream_t stream) {
    if (n_in < 67) return;
    (void)out_size;

    // ---- detect flattening order of the params pytree ----
    IdxMap ix;
    if (in_sizes[0] == NN * IN_DIM && in_sizes[4] == 256) {
        // full insertion order
        ix.x = 0; ix.ei = 1; ix.nt = 2; ix.et = 3; ix.nemb = 4; ix.eemb = 5;
        ix.inp = {6, 7, 8, 9, 10, 11, 12, 13};
        for (int L = 0; L < 3; ++L) {
            int b = 14 + 15 * L;
            ix.blk[L] = {b + 0, b + 1, b + 2, b + 3, b + 4, b + 5, b + 6,
                         {b + 7, b + 8, b + 9, b + 10, b + 11, b + 12, b + 13, b + 14}};
        }
        ix.out = {59, 60, 61, 62, 63, 64, 65, 66};
    } else if (in_sizes[0] == NN * IN_DIM) {
        // top-level insertion order, params dicts sorted alphabetically
        ix.x = 0; ix.ei = 1; ix.nt = 2; ix.et = 3;
        for (int L = 0; L < 3; ++L) {
            int b = 4 + 15 * L;   // We,Wl,Wr,att,bias,bl,br, mlp{W1,W2,b1,b2,bb1,bb2,g1,g2}
            ix.blk[L] = {b + 1, b + 5, b + 2, b + 6, b + 0, b + 3, b + 4,
                         {b + 13, b + 9, b + 7, b + 11, b + 14, b + 10, b + 8, b + 12}};
        }
        ix.eemb = 49;
        ix.inp  = { /*g1*/56, /*b1*/52, /*W1*/50, /*bb1*/54, /*g2*/57, /*b2*/53, /*W2*/51, /*bb2*/55 };
        ix.nemb = 58;
        ix.out  = { /*g1*/65, /*b1*/61, /*W1*/59, /*bb1*/63, /*g2*/66, /*b2*/62, /*W2*/60, /*bb2*/64 };
    } else {
        // fully sorted (jax tree_flatten default)
        ix.ei = 0; ix.et = 1; ix.nt = 2;
        for (int L = 0; L < 3; ++L) {
            int b = 3 + 15 * L;   // We,Wl,Wr,att,bias,bl,br, mlp{W1,W2,b1,b2,bb1,bb2,g1,g2}
            ix.blk[L] = {b + 1, b + 5, b + 2, b + 6, b + 0, b + 3, b + 4,
                         {b + 13, b + 9, b + 7, b + 11, b + 14, b + 10, b + 8, b + 12}};
        }
        ix.eemb = 48;
        ix.inp  = { /*g1*/55, /*b1*/51, /*W1*/49, /*bb1*/53, /*g2*/56, /*b2*/52, /*W2*/50, /*bb2*/54 };
        ix.nemb = 57;
        ix.out  = { /*g1*/64, /*b1*/60, /*W1*/58, /*bb1*/62, /*g2*/65, /*b2*/61, /*W2*/59, /*bb2*/63 };
        ix.x = 66;
    }

    auto F = [&](int i) { return (const float*)d_in[i]; };
    auto I = [&](int i) { return (const int*)d_in[i]; };

    const float* x     = F(ix.x);
    const int*   src   = I(ix.ei);
    const int*   dst   = I(ix.ei) + EE;
    const int*   ntype = I(ix.nt);
    const int*   etype = I(ix.et);
    const float* nemb  = F(ix.nemb);
    const float* eemb  = F(ix.eemb);

    // ---- workspace layout (float32 words, 256-word aligned regions) ----
    float* base = (float*)d_ws;
    size_t off = 0;
    auto alloc = [&](size_t n) { float* r = base + off; off += (n + 255) & ~(size_t)255; return r; };
    float*    h0    = alloc((size_t)NN * 160);   // reused as attention output (N x 128)
    float*    big   = alloc((size_t)NN * 256);   // hid160 (input phase) / MLP hidden (layers)
    float*    h     = alloc((size_t)NN * DD);
    float*    xl    = alloc((size_t)NN * DD);
    float*    xr    = alloc((size_t)NN * DD);
    float*    lat   = alloc((size_t)NN * DT);
    float*    cnt   = alloc((size_t)NN);
    float*    lee   = alloc((size_t)NN * DD);
    float*    esT   = alloc((size_t)16 * DD);
    float*    sbuf  = alloc((size_t)(EE + NN) * 8);
    unsigned* smax  = (unsigned*)alloc((size_t)NN * 8);
    float*    den   = alloc((size_t)NN * 8);
    float*    sums  = alloc(512);
    float*    Wf    = alloc((size_t)256 * 256);
    float*    bf    = alloc(256);
    float*    attn  = h0;                        // alias: h0 dead after input MLP
    if (ws_size < off * sizeof(float)) return;

    const size_t edgeThreads = (size_t)(EE + NN) * 8;
    const unsigned edgeBlocks = (unsigned)((edgeThreads + 255) / 256);
    auto bn = [&](const float* X, int M, int C, int ld) {
        zero_u32(stream, sums, 512);
        k_bn_stats<<<(unsigned)((M + 511) / 512), 256, 0, stream>>>(X, M, C, ld, sums, 512);
    };
    auto fold = [&](const float* W, const MlpIdx& m, bool first, int Nout, int Kin, int Kpad, float invM) {
        const float* g  = first ? F(m.g1)  : F(m.g2);
        const float* be = first ? F(m.b1)  : F(m.b2);
        const float* bl = first ? F(m.bb1) : F(m.bb2);
        k_bn_fold<<<Nout, 256, 0, stream>>>(W, g, be, bl, sums, invM, Wf, bf, Kin, Kpad);
    };

    // ---- self-loop edge attribute = mean of incoming edge embeddings ----
    zero_u32(stream, lat, (size_t)NN * DT);
    zero_u32(stream, cnt, (size_t)NN);
    k_loop_scatter<<<(unsigned)(((size_t)EE * DT + 255) / 256), 256, 0, stream>>>(dst, etype, eemb, lat, cnt);
    k_loop_fin<<<(unsigned)(((size_t)NN * DT + 255) / 256), 256, 0, stream>>>(lat, cnt);

    // ---- input MLP: h0 (N x 160, K padded) -> hid160 -> h (N x 128) ----
    k_build_h0<<<(unsigned)(((size_t)NN * 160 + 255) / 256), 256, 0, stream>>>(x, ntype, nemb, h0);
    bn(h0, NN, 132, 160);
    fold(F(ix.inp.W1), ix.inp, true, 132, 132, 160, 1.0f / NN);
    gemm(stream, 160, h0, 160, Wf, 160, 132, bf, big, 160, NN, 132, 10, 0.01f, 1, 0);
    bn(big, NN, 132, 160);
    fold(F(ix.inp.W2), ix.inp, false, 128, 132, 160, 1.0f / NN);
    gemm(stream, 160, big, 160, Wf, 160, 128, bf, h, 128, NN, 128, 8, 0.01f, 1, 0);

    // ---- 3 GATv2 blocks ----
    for (int L = 0; L < 3; ++L) {
        const BlkIdx& B = ix.blk[L];
        gemm(stream, 128, h, 128, F(B.Wl), 128, 128, F(B.bl), xl, 128, NN, 128, 8, 0.f, 0, 0);
        gemm(stream, 128, h, 128, F(B.Wr), 128, 128, F(B.br), xr, 128, NN, 128, 8, 0.f, 0, 0);
        gemm(stream, 32, eemb, 32, F(B.We), 32, 128, nullptr, esT, 128, 16, 128, 8, 0.f, 0, 0);
        gemm(stream, 32, lat, 32, F(B.We), 32, 128, nullptr, lee, 128, NN, 128, 8, 0.f, 0, 0);

        zero_u32(stream, smax, (size_t)NN * 8);
        zero_u32(stream, den,  (size_t)NN * 8);
        k_init_bias<<<(unsigned)(((size_t)NN * DD + 255) / 256), 256, 0, stream>>>(attn, F(B.bias));

        k_gat_scores<<<edgeBlocks, 256, 0, stream>>>(xl, xr, esT, lee, src, dst, etype, F(B.att), sbuf, smax);
        k_gat_exp<<<edgeBlocks, 256, 0, stream>>>(dst, sbuf, smax, den);
        k_gat_agg<<<edgeBlocks, 256, 0, stream>>>(xl, src, dst, sbuf, den, attn);

        // v_update MLP with residual fused into second GEMM
        bn(attn, NN, 128, 128);
        fold(F(B.mlp.W1), B.mlp, true, 256, 128, 128, 1.0f / NN);
        gemm(stream, 128, attn, 128, Wf, 128, 256, bf, big, 256, NN, 256, 16, 0.01f, 1, 0);
        bn(big, NN, 256, 256);
        fold(F(B.mlp.W2), B.mlp, false, 128, 256, 256, 1.0f / NN);
        gemm(stream, 256, big, 256, Wf, 256, 128, bf, h, 128, NN, 128, 8, 0.01f, 1, 1); // h += ...
    }

    // ---- output MLP on first 4278 rows ----
    float* out = (float*)d_out;
    bn(h, NOUTR, 128, 128);
    fold(F(ix.out.W1), ix.out, true, 256, 128, 128, 1.0f / NOUTR);
    gemm(stream, 128, h, 128, Wf, 128, 256, bf, big, 256, NOUTR, 256, 16, 0.01f, 1, 0);
    bn(big, NOUTR, 256, 256);
    fold(F(ix.out.W2), ix.out, false, 64, 256, 256, 1.0f / NOUTR);
    gemm(stream, 256, big, 256, Wf, 256, 64, bf, out, 64, NOUTR, 64, 4, 0.01f, 1, 0);
}
